// WTDRLoss_43800076485193
// MI455X (gfx1250) — matrix operand order repaired and checked by reading, not compiled, not run on hardware
//
#include <hip/hip_runtime.h>

// ---------------------------------------------------------------------------
// Types for CDNA5 WMMA (wave32, gfx1250)
// ---------------------------------------------------------------------------
typedef __attribute__((ext_vector_type(16))) __bf16 v16bf;
typedef __attribute__((ext_vector_type(8)))  __bf16 v8bf;
typedef __attribute__((ext_vector_type(8)))  float  v8f;

#define MARGIN 0.3f
#define EPSV   1e-9f

__device__ __forceinline__ unsigned short f2bf(float f) {
    unsigned int u = __float_as_uint(f);
    u += 0x7FFFu + ((u >> 16) & 1u);           // round-to-nearest-even
    return (unsigned short)(u >> 16);
}

// AS3 offset of a shared-memory pointer: generic LDS VA carries the group-
// segment offset in its low 32 bits (aperture in the high bits).
__device__ __forceinline__ unsigned lds_off(const void* p) {
    return (unsigned)(uintptr_t)p;
}

// GLOBAL_LOAD_ASYNC_TO_LDS_B128, GV mode: per-lane 16B global -> LDS,
// tracked by ASYNCcnt (CDNA5 ISA).
__device__ __forceinline__ void async_ld_b128(unsigned lds, const void* g) {
    asm volatile("global_load_async_to_lds_b128 %0, %1, off"
                 :: "v"(lds), "v"(g) : "memory");
}
__device__ __forceinline__ void wait_async0() {
    asm volatile("s_wait_asynccnt 0" ::: "memory");
}

// ---------------------------------------------------------------------------
// Kernel 0: fp32 -> bf16 conversion + exact fp32 row squared-norms.
// One 256-thread workgroup per row of 2048.
// ---------------------------------------------------------------------------
__global__ __launch_bounds__(256)
void convert_norm_kernel(const float* __restrict__ x,
                         unsigned short* __restrict__ xbf,
                         float* __restrict__ norms, int d)
{
    const int row = blockIdx.x;
    const float4* src = (const float4*)(x + (size_t)row * d);
    ushort4* dst = (ushort4*)(xbf + (size_t)row * d);
    float ss = 0.0f;
    for (int c = threadIdx.x; c < (d >> 2); c += blockDim.x) {
        float4 v = src[c];
        ss += v.x * v.x + v.y * v.y + v.z * v.z + v.w * v.w;
        ushort4 o;
        o.x = f2bf(v.x); o.y = f2bf(v.y); o.z = f2bf(v.z); o.w = f2bf(v.w);
        dst[c] = o;
    }
    #pragma unroll
    for (int off = 16; off > 0; off >>= 1) ss += __shfl_xor(ss, off, 32);
    __shared__ float wsum[8];
    if ((threadIdx.x & 31) == 0) wsum[threadIdx.x >> 5] = ss;
    __syncthreads();
    if (threadIdx.x == 0) {
        float t = 0.0f;
        #pragma unroll
        for (int w = 0; w < 8; ++w) t += wsum[w];
        norms[row] = t;
    }
}

// ---------------------------------------------------------------------------
// Kernel 1: init reduction buffers (fresh every launch -> deterministic).
// ---------------------------------------------------------------------------
__global__ void init_kernel(int* __restrict__ rowMaxBits, int* __restrict__ rowMinBits,
                            int* __restrict__ maxLBits, int* __restrict__ correctCnt, int n3)
{
    int i = blockIdx.x * blockDim.x + threadIdx.x;
    if (i < n3) {
        rowMaxBits[i] = __float_as_int(-1e30f);
        rowMinBits[i] = __float_as_int( 1e30f);
    }
    if (i < 3) maxLBits[i] = 0;
    if (i == 0) *correctCnt = 0;
}

// ---------------------------------------------------------------------------
// Kernel 2: fused bf16-WMMA GEMM (A @ B^T) -> distance -> masked row max/min.
// Block tile 128x128, 256 threads = 8 wave32 waves, each wave 64x32 via
// 4x2 v_wmma_f32_16x16x32_bf16 accumulators. BK = 32.
// Double-buffered LDS staged with global_load_async_to_lds_b128 so the
// global->LDS traffic overlaps the WMMA pipe; one barrier per K-step.
// LDS row stride 48 halves (96B = 6*16B) keeps b128 accesses aligned.
// ---------------------------------------------------------------------------
__global__ __launch_bounds__(256)
void gemm_dist_kernel(const unsigned short* __restrict__ xbf,
                      const float* __restrict__ norms,
                      const int* __restrict__ t,
                      int aBase, int bBase,
                      int* __restrict__ rowMaxBits,
                      int* __restrict__ rowMinBits,
                      int n, int d)
{
    constexpr int BK  = 32;
    constexpr int LDT = 48;                       // LDS row stride (halves)
    __shared__ unsigned short lA[2][128 * LDT];   // 2 x 12 KB
    __shared__ unsigned short lB[2][128 * LDT];   // 2 x 12 KB

    const int bm   = blockIdx.x * 128;
    const int bn   = blockIdx.y * 128;
    const int tid  = threadIdx.x;
    const int lane = tid & 31;
    const int wave = tid >> 5;
    const int wm   = (wave >> 2) * 64;            // 0 / 64
    const int wn   = (wave & 3) * 32;             // 0 / 32 / 64 / 96
    const int half = lane >> 4;                   // 0 / 1
    const int l16  = lane & 15;

    const v8f zero8 = {0.f, 0.f, 0.f, 0.f, 0.f, 0.f, 0.f, 0.f};
    v8f acc[4][2];
    #pragma unroll
    for (int mi = 0; mi < 4; ++mi)
        #pragma unroll
        for (int ni = 0; ni < 2; ++ni) acc[mi][ni] = zero8;

    // Tile = 128x32 halves = 512 sixteen-byte chunks; 2 chunks/thread/matrix.
    // chunk c -> row = c>>2 (0..127), seg = (c&3)*8 halves (16B).
    const int r0 = tid >> 2,         s0 = (tid & 3) * 8;
    const int r1 = (tid + 256) >> 2, s1 = ((tid + 256) & 3) * 8;

    auto issue = [&](int buf, int kk) {
        async_ld_b128(lds_off(&lA[buf][r0 * LDT + s0]),
                      xbf + (size_t)(aBase + bm + r0) * d + kk + s0);
        async_ld_b128(lds_off(&lA[buf][r1 * LDT + s1]),
                      xbf + (size_t)(aBase + bm + r1) * d + kk + s1);
        async_ld_b128(lds_off(&lB[buf][r0 * LDT + s0]),
                      xbf + (size_t)(bBase + bn + r0) * d + kk + s0);
        async_ld_b128(lds_off(&lB[buf][r1 * LDT + s1]),
                      xbf + (size_t)(bBase + bn + r1) * d + kk + s1);
    };

    const int S = d / BK;
    issue(0, 0);
    wait_async0();
    __syncthreads();

    for (int s = 0; s < S; ++s) {
        const int p = s & 1;
        if (s + 1 < S) issue(p ^ 1, (s + 1) * BK);   // prefetch next tile

        // A fragment (16x32 bf16): lane m=l16; halves 0/1 own K bases 0/8,
        // vector elems [0..7]=K kb..kb+7, [8..15]=K kb+16..kb+23.
        v16bf aF[4], bF[2];
        const int akb = half * 8;
        #pragma unroll
        for (int mi = 0; mi < 4; ++mi) {
            int m = wm + mi * 16 + l16;
            v8bf lo = *(const v8bf*)&lA[p][m * LDT + akb];
            v8bf hi = *(const v8bf*)&lA[p][m * LDT + akb + 16];
            aF[mi] = __builtin_shufflevector(lo, hi,
                     0, 1, 2, 3, 4, 5, 6, 7, 8, 9, 10, 11, 12, 13, 14, 15);
        }
        // B fragment (32x16 bf16): lane n=l16; halves own K bases 0/8, same
        // order as A, stored N-major (rows of the right input matrix).
        const int bkb = half * 8;
        #pragma unroll
        for (int ni = 0; ni < 2; ++ni) {
            int nr = wn + ni * 16 + l16;
            v8bf lo = *(const v8bf*)&lB[p][nr * LDT + bkb];
            v8bf hi = *(const v8bf*)&lB[p][nr * LDT + bkb + 16];
            bF[ni] = __builtin_shufflevector(lo, hi,
                     0, 1, 2, 3, 4, 5, 6, 7, 8, 9, 10, 11, 12, 13, 14, 15);
        }
        #pragma unroll
        for (int mi = 0; mi < 4; ++mi)
            #pragma unroll
            for (int ni = 0; ni < 2; ++ni)
                acc[mi][ni] = __builtin_amdgcn_wmma_f32_16x16x32_bf16(
                    false, aF[mi], false, bF[ni], (short)0, acc[mi][ni],
                    false, false);

        if (s + 1 < S) {
            wait_async0();     // own share of next tile landed
            __syncthreads();   // whole tile visible; buf p safe to overwrite
        }
    }

    // ---- Fused epilogue: distance + masked row max / unmasked row min ----
    // C layout: lane -> col = l16; VGPR r -> row = r + 8*half (per 16x16 tile)
    float nbv[2]; int tjv[2];
    #pragma unroll
    for (int ni = 0; ni < 2; ++ni) {
        int j = bn + wn + ni * 16 + l16;
        nbv[ni] = norms[bBase + j];
        tjv[ni] = t[j];
    }
    #pragma unroll
    for (int mi = 0; mi < 4; ++mi) {
        #pragma unroll
        for (int r = 0; r < 8; ++r) {
            int   i   = bm + wm + mi * 16 + r + 8 * half;
            float nai = norms[aBase + i];
            int   ti  = t[i];
            float vmax = -1e30f, vmin = 1e30f;
            #pragma unroll
            for (int ni = 0; ni < 2; ++ni) {
                float c  = acc[mi][ni][r];
                float dd = sqrtf(fmaxf(nai + nbv[ni] - 2.0f * c, 1e-12f));
                if (ti == tjv[ni]) vmax = fmaxf(vmax, dd);
                else               vmin = fminf(vmin, dd);
            }
            #pragma unroll
            for (int off = 1; off < 16; off <<= 1) {
                vmax = fmaxf(vmax, __shfl_xor(vmax, off, 32));
                vmin = fminf(vmin, __shfl_xor(vmin, off, 32));
            }
            // positive floats (and the +-1e30 sentinels) order correctly
            // under signed-int compare -> bit-pattern atomics are exact.
            if (l16 == 0) {
                atomicMax(rowMaxBits + i, __float_as_int(vmax));
                atomicMin(rowMinBits + i, __float_as_int(vmin));
            }
        }
    }
}

// ---------------------------------------------------------------------------
// Kernel 3: per-row losses l1/l2/l3 -> global max(l_k) + correct count.
// max and integer add are order-independent -> deterministic.
// ---------------------------------------------------------------------------
__global__ __launch_bounds__(256)
void mine_reduce_kernel(const int* __restrict__ mxb, const int* __restrict__ mnb,
                        int* __restrict__ maxLBits, int* __restrict__ correctCnt, int n)
{
    int i = blockIdx.x * blockDim.x + threadIdx.x;
    float l1 = 0.f, l2 = 0.f, l3 = 0.f; int c = 0;
    if (i < n) {
        float ap1 = __int_as_float(mxb[i]);
        float ap2 = __int_as_float(mxb[n + i]);
        float ap3 = __int_as_float(mxb[2 * n + i]);
        float an1 = __int_as_float(mnb[2 * n + i]);   // mine(dist1, dist3)
        float an2 = __int_as_float(mnb[i]);           // mine(dist2, dist1)
        float an3 = __int_as_float(mnb[n + i]);       // mine(dist3, dist2)
        l1 = fmaxf(0.f, ap1 - an1 + MARGIN);
        l2 = fmaxf(0.f, ap2 - an2 + MARGIN);
        l3 = fmaxf(0.f, ap3 - an3 + MARGIN);
        c  = (an1 >= ap1) + (an2 >= ap2) + (an3 >= ap3);
    }
    #pragma unroll
    for (int off = 16; off > 0; off >>= 1) {
        l1 = fmaxf(l1, __shfl_xor(l1, off, 32));
        l2 = fmaxf(l2, __shfl_xor(l2, off, 32));
        l3 = fmaxf(l3, __shfl_xor(l3, off, 32));
        c += __shfl_xor(c, off, 32);
    }
    if ((threadIdx.x & 31) == 0) {
        atomicMax(maxLBits + 0, __float_as_int(l1));
        atomicMax(maxLBits + 1, __float_as_int(l2));
        atomicMax(maxLBits + 2, __float_as_int(l3));
        atomicAdd(correctCnt, c);
    }
}

// ---------------------------------------------------------------------------
// Kernel 4: weighted-loss sum (single block, fixed tree -> deterministic)
// and final output write: out[0]=loss, out[1]=correct.
// ---------------------------------------------------------------------------
__global__ __launch_bounds__(256)
void weight_sum_kernel(const int* __restrict__ mxb, const int* __restrict__ mnb,
                       const int* __restrict__ maxLBits, const int* __restrict__ correctCnt,
                       float* __restrict__ out, int n)
{
    const float m1 = __int_as_float(maxLBits[0]) + EPSV;
    const float m2 = __int_as_float(maxLBits[1]) + EPSV;
    const float m3 = __int_as_float(maxLBits[2]) + EPSV;
    float s = 0.f;
    for (int i = threadIdx.x; i < n; i += blockDim.x) {
        float ap1 = __int_as_float(mxb[i]);
        float ap2 = __int_as_float(mxb[n + i]);
        float ap3 = __int_as_float(mxb[2 * n + i]);
        float an1 = __int_as_float(mnb[2 * n + i]);
        float an2 = __int_as_float(mnb[i]);
        float an3 = __int_as_float(mnb[n + i]);
        float l1 = fmaxf(0.f, ap1 - an1 + MARGIN);
        float l2 = fmaxf(0.f, ap2 - an2 + MARGIN);
        float l3 = fmaxf(0.f, ap3 - an3 + MARGIN);
        float w1 = l1 / m1, w2 = l2 / m2, w3 = l3 / m3;
        float H  = 3.f * w1 * w2 * w3 / (w1 * w3 + w2 * w3 + w1 * w2 + EPSV);
        s += (1.f - (w1 + EPSV) * H) * l1
           + (1.f - (w2 + EPSV) * H) * l2
           + (1.f - (w3 + EPSV) * H) * l3;      // GAMMA == 1
    }
    __shared__ float red[256];
    red[threadIdx.x] = s;
    __syncthreads();
    #pragma unroll
    for (int off = 128; off > 0; off >>= 1) {
        if (threadIdx.x < off) red[threadIdx.x] += red[threadIdx.x + off];
        __syncthreads();
    }
    if (threadIdx.x == 0) {
        out[0] = red[0] / (3.0f * (float)n);
        out[1] = (float)(*correctCnt);
    }
}

// ---------------------------------------------------------------------------
// Host launch
// ---------------------------------------------------------------------------
extern "C" void kernel_launch(void* const* d_in, const int* in_sizes, int n_in,
                              void* d_out, int out_size, void* d_ws, size_t ws_size,
                              hipStream_t stream)
{
    const float* inputs  = (const float*)d_in[0];
    const int*   targets = (const int*)d_in[1];
    // d_in[2] (m_targets = arange(2n)//n) is already sorted -> identity order.

    const int n = in_sizes[1] / 3;             // 4096
    const int d = in_sizes[0] / in_sizes[1];   // 2048

    // Workspace layout
    unsigned short* xbf   = (unsigned short*)d_ws;               // 3n*d bf16
    float* norms          = (float*)((char*)d_ws + (size_t)3 * n * d * sizeof(unsigned short));
    int*   rowMaxBits     = (int*)(norms + (size_t)3 * n);       // 3 arrays of n
    int*   rowMinBits     = rowMaxBits + (size_t)3 * n;          // 3 arrays of n
    int*   maxLBits       = rowMinBits + (size_t)3 * n;          // 3 scalars
    int*   correctCnt     = maxLBits + 3;                        // 1 scalar

    const int* t = targets + 2 * n;            // mask targets

    // Phase 0: convert + norms
    convert_norm_kernel<<<3 * n, 256, 0, stream>>>(inputs, xbf, norms, d);

    // Phase 1: init reductions
    init_kernel<<<(3 * n + 255) / 256, 256, 0, stream>>>(rowMaxBits, rowMinBits,
                                                         maxLBits, correctCnt, 3 * n);

    // Phase 2: three fused GEMM+mine passes
    dim3 grid(n / 128, n / 128);
    // dist1 = pdist(input1, input2): rows [0,n), cols [2n,3n)
    gemm_dist_kernel<<<grid, 256, 0, stream>>>(xbf, norms, t, 0,     2 * n,
                                               rowMaxBits,         rowMinBits,         n, d);
    // dist2 = pdist(input2, input3): rows [2n,3n), cols [n,2n)
    gemm_dist_kernel<<<grid, 256, 0, stream>>>(xbf, norms, t, 2 * n, n,
                                               rowMaxBits + n,     rowMinBits + n,     n, d);
    // dist3 = pdist(input1, input3): rows [0,n), cols [n,2n)
    gemm_dist_kernel<<<grid, 256, 0, stream>>>(xbf, norms, t, 0,     n,
                                               rowMaxBits + 2 * n, rowMinBits + 2 * n, n, d);

    // Phase 3: per-row losses -> max(l_k), correct count
    mine_reduce_kernel<<<(n + 255) / 256, 256, 0, stream>>>(rowMaxBits, rowMinBits,
                                                            maxLBits, correctCnt, n);

    // Phase 4: deterministic weighted sum + output
    weight_sum_kernel<<<1, 256, 0, stream>>>(rowMaxBits, rowMinBits, maxLBits,
                                             correctCnt, (float*)d_out, n);
}
